// DotProductAttention2_86904368267444
// MI455X (gfx1250) — matrix operand order repaired and checked by reading, not compiled, not run on hardware
//
#include <hip/hip_runtime.h>
#include <stdint.h>

typedef __bf16 bf16;
typedef bf16  v16bf __attribute__((ext_vector_type(16)));
typedef float v8f   __attribute__((ext_vector_type(8)));

#define B_DIM 32
#define Q_DIM 2048
#define K_DIM 2048
#define D_DIM 128
#define BQ 64                    // Q rows per workgroup
#define BK 64                    // K cols per iteration
#define NW 4                     // waves per workgroup
#define NTHREADS (NW * 32)
#define SP_STRIDE (BK + 8)       // 72 bf16 -> 144B rows; b128 reads conflict-free
// scores are computed in the exp2 domain: Q pre-scaled by (1/sqrt(D))*log2(e)
#define QSCALE 0.1275303283786536f        // 0.08838834764831845 * 1.4426950408889634
#define MASK2  (-1442695.0408889634f)     // -1e6 * log2(e)

union Frag {
  v16bf v;
  uint4 q[2];
  bf16  h[16];
};

// ---------------- pre-pass 1: elementwise f32 -> bf16 (optional scale) ----------------
__global__ __launch_bounds__(256) void cvt_bf16(const float* __restrict__ src,
                                                bf16* __restrict__ dst,
                                                float scale, int n4) {
  int i = blockIdx.x * 256 + threadIdx.x;
  if (i >= n4) return;
  float4 f = ((const float4*)src)[i];
  union { bf16 h[4]; uint2 u; } o;
  o.h[0] = (bf16)(f.x * scale);
  o.h[1] = (bf16)(f.y * scale);
  o.h[2] = (bf16)(f.z * scale);
  o.h[3] = (bf16)(f.w * scale);
  ((uint2*)dst)[i] = o.u;
}

// ---------------- pre-pass 2: V [b][k][d] f32 -> Vt [b][d][k] bf16 ----------------
__global__ __launch_bounds__(256) void transpose_v(const float* __restrict__ V,
                                                   bf16* __restrict__ Vt) {
  __shared__ __align__(16) bf16 sT[64][72];
  const int tpb = (K_DIM / 64) * (D_DIM / 64);   // tiles per batch = 32*2
  int id = blockIdx.x;
  int b  = id / tpb;
  int r  = id % tpb;
  int kt = r / (D_DIM / 64);
  int dt = r % (D_DIM / 64);
  int t  = threadIdx.x;
  int tr = t >> 4;          // 0..15
  int tc = t & 15;          // 0..15
#pragma unroll
  for (int p = 0; p < 4; ++p) {
    int row = p * 16 + tr;  // k within tile
    const float4* src = (const float4*)(V + ((size_t)b * K_DIM + kt * 64 + row) * D_DIM
                                          + dt * 64 + tc * 4);
    float4 f = *src;
    bf16* d = &sT[row][tc * 4];
    d[0] = (bf16)f.x; d[1] = (bf16)f.y; d[2] = (bf16)f.z; d[3] = (bf16)f.w;
  }
  __syncthreads();
#pragma unroll
  for (int p = 0; p < 4; ++p) {
    int dr = p * 16 + tr;   // d within tile
    int kc = tc * 4;        // k within tile
    union { bf16 h[4]; uint2 u; } o;
    o.h[0] = sT[kc + 0][dr];
    o.h[1] = sT[kc + 1][dr];
    o.h[2] = sT[kc + 2][dr];
    o.h[3] = sT[kc + 3][dr];
    *(uint2*)(Vt + ((size_t)b * D_DIM + dt * 64 + dr) * K_DIM + kt * 64 + kc) = o.u;
  }
}

// ---------------- main flash-attention kernel (bf16 WMMA, fragments from global) ----------------
// S-tile column permutation: score column of tile kt at lane ln corresponds to
// k_global = kb*BK + 4*ln + kt. Softmax is column-permutation invariant; P is
// stored into sP at its natural k position (col 4*ln + kt), so the P*V
// contraction ordering matches the contiguous Vt fragments exactly.
__global__ __launch_bounds__(NTHREADS) void
fattn_bf16_wmma(const bf16* __restrict__ Qbf,   // pre-scaled by QSCALE
                const bf16* __restrict__ Kbf,   // [b][k][d]
                const bf16* __restrict__ Vt,    // [b][d][k]
                const uint8_t* __restrict__ Mg,
                float* __restrict__ Og) {
  __shared__ __align__(16) bf16 sP[NW * 16 * SP_STRIDE];   // per-wave P scratch only

  const int tid  = threadIdx.x;
  const int w    = tid >> 5;
  const int lane = tid & 31;
  const int h    = lane >> 4;
  const int ln   = lane & 15;

  const int b    = blockIdx.x / (Q_DIM / BQ);
  const int qblk = (blockIdx.x % (Q_DIM / BQ)) * BQ;
  const int qw   = qblk + w * 16;

  // A fragment of Q (16x128 in 4 chunks of K=32); layout per ISA 7.12.2
  Frag qf[4];
  {
    const bf16* qrow = Qbf + ((size_t)b * Q_DIM + (qw + ln)) * D_DIM;
#pragma unroll
    for (int c = 0; c < 4; ++c) {
      qf[c].q[0] = *(const uint4*)(qrow + 32 * c + 8 * h);
      qf[c].q[1] = *(const uint4*)(qrow + 32 * c + 16 + 8 * h);
    }
  }

  v8f   Ot[8];
  float mrow[8], lrow[8];           // lrow is a per-lane PARTIAL sum (reduced at end)
#pragma unroll
  for (int t = 0; t < 8; ++t)
#pragma unroll
    for (int r = 0; r < 8; ++r) Ot[t][r] = 0.0f;
#pragma unroll
  for (int r = 0; r < 8; ++r) { mrow[r] = -3.0e37f; lrow[r] = 0.0f; }

  const uint8_t* mrowbase = Mg + ((size_t)b * Q_DIM + qw) * (size_t)K_DIM;
  const bf16* kbatch = Kbf + (size_t)b * K_DIM * D_DIM;
  const bf16* vbatch = Vt  + (size_t)b * D_DIM * K_DIM;

  for (int kb = 0; kb < K_DIM / BK; ++kb) {
    // prefetch next K/V block through L2 (global_prefetch_b8)
    if (kb + 1 < K_DIM / BK) {
      const bf16* knext = kbatch + (size_t)(kb + 1) * BK * D_DIM;
      __builtin_prefetch(knext + lane * 256, 0, 1);
      __builtin_prefetch(knext + 8192 + lane * 256, 0, 1);
#pragma unroll
      for (int j = 0; j < 4; ++j)
        __builtin_prefetch(vbatch + (size_t)(j * 32 + lane) * K_DIM + (kb + 1) * BK, 0, 1);
    }

    // ---- S = Q * K^T : four 16x16 k-tiles (permuted cols), D in 4 chunks of 32 ----
    // B frag: lane col ln of tile kt holds K row (4*ln + kt); elems are d = 32c+8h+0..7, 32c+16+8h+0..7.
    v8f s[4];
#pragma unroll
    for (int kt = 0; kt < 4; ++kt)
#pragma unroll
      for (int r = 0; r < 8; ++r) s[kt][r] = 0.0f;
#pragma unroll
    for (int kt = 0; kt < 4; ++kt) {
      const bf16* krow = kbatch + (size_t)(kb * BK + 4 * ln + kt) * D_DIM + h * 8;
#pragma unroll
      for (int c = 0; c < 4; ++c) {
        Frag bk;
        bk.q[0] = *(const uint4*)(krow + c * 32);
        bk.q[1] = *(const uint4*)(krow + c * 32 + 16);
        s[kt] = __builtin_amdgcn_wmma_f32_16x16x32_bf16(
            false, qf[c].v, false, bk.v, (short)0, s[kt], false, false);
      }
    }

    // ---- mask (one u32/row) + online softmax in exp2 domain ----
#pragma unroll
    for (int r = 0; r < 8; ++r) {
      size_t moff = (size_t)(r + 8 * h) * K_DIM + (size_t)kb * BK + 4 * ln;
      uint32_t mu = *(const uint32_t*)(mrowbase + moff);
      float v0 = s[0][r];
      float v1 = s[1][r];
      float v2 = s[2][r];
      float v3 = s[3][r];
      if (mu & 0x000000ffu) v0 = MASK2;
      if (mu & 0x0000ff00u) v1 = MASK2;
      if (mu & 0x00ff0000u) v2 = MASK2;
      if (mu & 0xff000000u) v3 = MASK2;

      float vmax = fmaxf(fmaxf(v0, v1), fmaxf(v2, v3));
      vmax = fmaxf(vmax, __shfl_xor(vmax, 1, 32));
      vmax = fmaxf(vmax, __shfl_xor(vmax, 2, 32));
      vmax = fmaxf(vmax, __shfl_xor(vmax, 4, 32));
      vmax = fmaxf(vmax, __shfl_xor(vmax, 8, 32));

      float mnew = fmaxf(mrow[r], vmax);
      float corr = exp2f(mrow[r] - mnew);   // row-uniform across lanes
      float e0 = exp2f(v0 - mnew);
      float e1 = exp2f(v1 - mnew);
      float e2 = exp2f(v2 - mnew);
      float e3 = exp2f(v3 - mnew);
      // per-lane partial row sum; cross-lane reduction deferred to epilogue
      lrow[r] = lrow[r] * corr + ((e0 + e1) + (e2 + e3));
      mrow[r] = mnew;
#pragma unroll
      for (int t = 0; t < 8; ++t) Ot[t][r] *= corr;

      // P at natural k position: cols 4*ln .. 4*ln+3 -> one 8B store
      union { bf16 hh[4]; uint2 u; } pe;
      pe.hh[0] = (bf16)e0;
      pe.hh[1] = (bf16)e1;
      pe.hh[2] = (bf16)e2;
      pe.hh[3] = (bf16)e3;
      *(uint2*)(&sP[(w * 16 + r + 8 * h) * SP_STRIDE + 4 * ln]) = pe.u;
    }

    // ---- reload P (16x64) as two A fragments (wave-private LDS, in-order DS) ----
    Frag pf[2];
#pragma unroll
    for (int c = 0; c < 2; ++c) {
      const bf16* pb = &sP[(w * 16 + ln) * SP_STRIDE + 32 * c + 8 * h];
      pf[c].q[0] = *(const uint4*)pb;
      pf[c].q[1] = *(const uint4*)(pb + 16);
    }

    // ---- O += P * V over 8 d-tiles x 2 k-chunks; B frag from transposed Vt ----
#pragma unroll
    for (int t = 0; t < 8; ++t) {
      const bf16* vrow = vbatch + (size_t)(t * 16 + ln) * K_DIM + kb * BK + 8 * h;
#pragma unroll
      for (int c = 0; c < 2; ++c) {
        Frag bv;
        bv.q[0] = *(const uint4*)(vrow + c * 32);
        bv.q[1] = *(const uint4*)(vrow + c * 32 + 16);
        Ot[t] = __builtin_amdgcn_wmma_f32_16x16x32_bf16(
            false, pf[c].v, false, bv.v, (short)0, Ot[t], false, false);
      }
    }
  }

  // ---- final cross-lane reduction of l, normalize, store ----
  float* obase = Og + ((size_t)b * Q_DIM + qw) * D_DIM;
#pragma unroll
  for (int r = 0; r < 8; ++r) {
    float l = lrow[r];
    l += __shfl_xor(l, 1, 32);
    l += __shfl_xor(l, 2, 32);
    l += __shfl_xor(l, 4, 32);
    l += __shfl_xor(l, 8, 32);
    float inv = 1.0f / l;
    size_t ro = (size_t)(r + 8 * h) * D_DIM;
#pragma unroll
    for (int t = 0; t < 8; ++t)
      obase[ro + t * 16 + ln] = Ot[t][r] * inv;
  }
}

extern "C" void kernel_launch(void* const* d_in, const int* in_sizes, int n_in,
                              void* d_out, int out_size, void* d_ws, size_t ws_size,
                              hipStream_t stream) {
  const float*   q = (const float*)d_in[0];
  const float*   k = (const float*)d_in[1];
  const float*   v = (const float*)d_in[2];
  const uint8_t* m = (const uint8_t*)d_in[3];
  float* out = (float*)d_out;

  const size_t elems = (size_t)B_DIM * Q_DIM * D_DIM;       // 8388608 (Q==K dims)
  const size_t bytes = elems * sizeof(bf16);                // 16 MB each
  if (ws_size < 3 * bytes) return;                          // need ~50.3 MB scratch

  bf16* qbf = (bf16*)d_ws;
  bf16* kbf = qbf + elems;
  bf16* vt  = kbf + elems;

  const int n4 = (int)(elems / 4);
  dim3 cgrid((n4 + 255) / 256), cblk(256);
  cvt_bf16<<<cgrid, cblk, 0, stream>>>(q, qbf, QSCALE, n4);
  cvt_bf16<<<cgrid, cblk, 0, stream>>>(k, kbf, 1.0f, n4);

  dim3 tgrid(B_DIM * (K_DIM / 64) * (D_DIM / 64)), tblk(256);
  transpose_v<<<tgrid, tblk, 0, stream>>>(v, vt);

  dim3 grid(B_DIM * (Q_DIM / BQ)), block(NTHREADS);
  fattn_bf16_wmma<<<grid, block, 0, stream>>>(qbf, kbf, vt, m, out);
}
